// GripNetSuperEdges_6416681140880
// MI455X (gfx1250) — matrix operand order repaired and checked by reading, not compiled, not run on hardware
//
#include <hip/hip_runtime.h>
#include <hip/hip_bf16.h>

typedef __attribute__((ext_vector_type(16))) _Float16 v16h;
typedef __attribute__((ext_vector_type(8)))  float    v8f;
typedef __attribute__((ext_vector_type(4)))  float    v4f;

#define D_FEAT 128
#define TFD    32

// ---------------------------------------------------------------------------
// Init: accum[N_TARGET*128] = 0, deg[N_TARGET] = 1.0 (the "+1" self-loop term)
// ---------------------------------------------------------------------------
__global__ __launch_bounds__(256)
void init_kernel(float* __restrict__ accum, float* __restrict__ deg, int n_target) {
    int i = blockIdx.x * blockDim.x + threadIdx.x;
    int n_acc = n_target * D_FEAT;
    if (i < n_acc) {
        accum[i] = 0.0f;
    } else if (i < n_acc + n_target) {
        deg[i - n_acc] = 1.0f;
    }
}

// ---------------------------------------------------------------------------
// Pre-pack W (f32, row-major 128x128) into f16 laid out exactly as the WMMA
// B-matrix register image, so each lane's 16-element fragment is one
// contiguous 32-byte load.
// Fragment index: [kk = k0/32 (0..3)][g = n0/16 (0..7)][lane (0..31)][j (0..15)]
// maps to W[kk*32 + (lane>>4)*8 + (j<8 ? j : j+8), g*16 + (lane&15)].
// ---------------------------------------------------------------------------
__global__ __launch_bounds__(256)
void pack_w_kernel(const float* __restrict__ W, _Float16* __restrict__ Wpk) {
    int tid = blockIdx.x * blockDim.x + threadIdx.x;   // 0 .. 16383
    if (tid >= 4 * 8 * 32 * 16) return;
    int j    = tid & 15;
    int lane = (tid >> 4) & 31;
    int g    = (tid >> 9) & 7;
    int kk   = tid >> 12;
    int k = kk * 32 + (lane >> 4) * 8 + (j < 8 ? j : j + 8);
    int n = g * 16 + (lane & 15);
    Wpk[tid] = (_Float16)W[(size_t)k * D_FEAT + n];
}

// ---------------------------------------------------------------------------
// GEMM h = x @ W via V_WMMA_F32_16X16X32_F16.
// One wave -> one 16x16 output tile. Block = 8 waves = one 16-row strip
// across all 128 columns. K=128 -> 4 WMMA steps of K=32.
// x is streamed once -> non-temporal loads (keep L2 for h/accum reuse).
// ---------------------------------------------------------------------------
__global__ __launch_bounds__(256)
void gemm_wmma_kernel(const float* __restrict__ x, const _Float16* __restrict__ Wpk,
                      float* __restrict__ h, int M) {
    const int wave = threadIdx.x >> 5;
    const int lane = threadIdx.x & 31;
    const int m16  = lane & 15;          // 0..15
    const int hi   = lane >> 4;          // 0 or 1 (lane group)
    const int row0 = blockIdx.x * 16;
    const int n0   = wave * 16;

    int mrow  = row0 + m16;
    int mload = mrow < M ? mrow : (M - 1);   // clamp reads for ragged tail

    const float* xrow = x + (size_t)mload * D_FEAT;

    v8f c = {};
    #pragma unroll
    for (int kk = 0; kk < 4; ++kk) {
        const int k0 = kk * 32;
        // A tile 16x32 f16: lane(m16,hi): element j -> K = k0 + hi*8 + (j<8 ? j : j+8)
        const v4f* xv = (const v4f*)(xrow + k0 + hi * 8);
        v4f x0 = __builtin_nontemporal_load(xv + 0);
        v4f x1 = __builtin_nontemporal_load(xv + 1);
        v4f x2 = __builtin_nontemporal_load(xv + 4);   // +16 floats
        v4f x3 = __builtin_nontemporal_load(xv + 5);
        v16h a;
        #pragma unroll
        for (int j = 0; j < 4; ++j) {
            a[j]      = (_Float16)x0[j];
            a[4 + j]  = (_Float16)x1[j];
            a[8 + j]  = (_Float16)x2[j];
            a[12 + j] = (_Float16)x3[j];
        }

        // B fragment: one contiguous 32B load from the pre-packed image
        v16h b = *(const v16h*)(Wpk + (((size_t)kk * 8 + wave) * 32 + lane) * 16);

        c = __builtin_amdgcn_wmma_f32_16x16x32_f16(
                /*neg_a=*/false, a, /*neg_b=*/false, b,
                /*c_mod=*/(short)0, c, /*reuse_a=*/false, /*reuse_b=*/false);
    }

    // C/D layout: VGPR r -> M = row0 + r + 8*hi, N = n0 + m16
    #pragma unroll
    for (int r = 0; r < 8; ++r) {
        int mr = row0 + r + 8 * hi;
        if (mr < M) h[(size_t)mr * D_FEAT + n0 + m16] = c[r];
    }
}

// ---------------------------------------------------------------------------
// deg[row[e]] += ew[e]  (thread per edge; deg region is tiny -> L2 resident)
// ---------------------------------------------------------------------------
__global__ __launch_bounds__(256)
void degree_kernel(const int* __restrict__ row, const float* __restrict__ ew,
                   float* __restrict__ deg, int E) {
    int e = blockIdx.x * blockDim.x + threadIdx.x;
    if (e < E) unsafeAtomicAdd(&deg[row[e]], ew[e]);
}

// deg -> rsqrt(deg) in place (deg >= 1 always)
__global__ __launch_bounds__(256)
void rsqrt_kernel(float* __restrict__ deg, int n) {
    int i = blockIdx.x * blockDim.x + threadIdx.x;
    if (i < n) deg[i] = rsqrtf(deg[i]);
}

// ---------------------------------------------------------------------------
// Scatter: accum[col[e], :] += (dinv[row[e]] * ew[e]) * h[row[e], :]
// One wave per edge, 4 fp32 atomics per lane (coalesced 128-float row).
// h[0:50000] and accum both fit in L2 (192MB) -> L2-atomic bound.
// dinv at the shifted target nodes is exactly 1.0 (no incoming row edges).
// ---------------------------------------------------------------------------
__global__ __launch_bounds__(256)
void scatter_kernel(const float* __restrict__ h, const float* __restrict__ dinv,
                    const float* __restrict__ ew,
                    const int* __restrict__ row, const int* __restrict__ col,
                    float* __restrict__ accum, int E) {
    int e = blockIdx.x * (blockDim.x >> 5) + (threadIdx.x >> 5);
    if (e >= E) return;
    int lane = threadIdx.x & 31;

    int r = row[e];
    int c = col[e];
    float norm = dinv[r] * ew[e];            // * dinv[col]==1.0

    const float* hp = h     + (size_t)r * D_FEAT;
    float*       ap = accum + (size_t)c * D_FEAT;
    #pragma unroll
    for (int i = 0; i < 4; ++i) {
        int d = lane + 32 * i;
        unsafeAtomicAdd(&ap[d], norm * hp[d]);
    }
}

// ---------------------------------------------------------------------------
// out[t, 0:128]   = relu(accum[t,:] + bias)
// out[t, 128:160] = |target_feat[t,:]|        (row stride 160)
// Output is write-once -> non-temporal stores.
// ---------------------------------------------------------------------------
__global__ __launch_bounds__(256)
void finalize_kernel(const float* __restrict__ accum, const float* __restrict__ bias,
                     const float* __restrict__ tf, float* __restrict__ out,
                     int n_target) {
    int i = blockIdx.x * blockDim.x + threadIdx.x;
    int total = n_target * (D_FEAT + TFD);
    if (i >= total) return;
    int t = i / (D_FEAT + TFD);
    int j = i - t * (D_FEAT + TFD);
    float v;
    if (j < D_FEAT) {
        v = fmaxf(accum[(size_t)t * D_FEAT + j] + bias[j], 0.0f);
    } else {
        v = fabsf(tf[(size_t)t * TFD + (j - D_FEAT)]);
    }
    __builtin_nontemporal_store(v, out + i);
}

extern "C" void kernel_launch(void* const* d_in, const int* in_sizes, int n_in,
                              void* d_out, int out_size, void* d_ws, size_t ws_size,
                              hipStream_t stream) {
    const float* x    = (const float*)d_in[0];   // [M, 128]
    const float* W    = (const float*)d_in[1];   // [128, 128]
    const float* bias = (const float*)d_in[2];   // [128]
    const float* tf   = (const float*)d_in[3];   // [n_target, 32]
    const float* ew   = (const float*)d_in[4];   // [E]
    const int*   eidx = (const int*)d_in[5];     // [2, E] flat

    const int M        = in_sizes[0] / D_FEAT;   // 100000
    const int n_target = in_sizes[3] / TFD;      // 50000
    const int E        = in_sizes[4];            // 640000

    const int* row = eidx;        // source ids (< n_target by construction)
    const int* col = eidx + E;    // target ids (unshifted == col - n_source)

    // workspace layout (256B-aligned regions)
    char* ws = (char*)d_ws;
    size_t off = 0;
    float* h     = (float*)(ws + off); off += ((size_t)M * D_FEAT * 4 + 255) & ~(size_t)255;
    float* accum = (float*)(ws + off); off += ((size_t)n_target * D_FEAT * 4 + 255) & ~(size_t)255;
    float* deg   = (float*)(ws + off); off += ((size_t)n_target * 4 + 255) & ~(size_t)255;
    _Float16* Wpk = (_Float16*)(ws + off);       // 16384 f16 = 32KB

    float* out = (float*)d_out;

    // 1) zero accumulators, deg = 1.0
    {
        int n = n_target * D_FEAT + n_target;
        init_kernel<<<(n + 255) / 256, 256, 0, stream>>>(accum, deg, n_target);
    }
    // 1b) pack W into f16 B-register layout (16384 elements)
    pack_w_kernel<<<64, 256, 0, stream>>>(W, Wpk);
    // 2) h = x @ W  (WMMA)
    {
        int blocks = (M + 15) / 16;
        gemm_wmma_kernel<<<blocks, 256, 0, stream>>>(x, Wpk, h, M);
    }
    // 3) deg += scatter(ew over row)
    degree_kernel<<<(E + 255) / 256, 256, 0, stream>>>(row, ew, deg, E);
    // 4) deg -> rsqrt(deg)
    rsqrt_kernel<<<(n_target + 255) / 256, 256, 0, stream>>>(deg, n_target);
    // 5) message scatter-add (wave per edge)
    {
        int waves_per_block = 256 / 32;
        int blocks = (E + waves_per_block - 1) / waves_per_block;
        scatter_kernel<<<blocks, 256, 0, stream>>>(h, deg, ew, row, col, accum, E);
    }
    // 6) relu + bias + concat |target_feat|
    {
        int n = n_target * (D_FEAT + TFD);
        finalize_kernel<<<(n + 255) / 256, 256, 0, stream>>>(accum, bias, tf, out, n_target);
    }
}
